// CriticGCN_1503238553877
// MI455X (gfx1250) — matrix operand order repaired and checked by reading, not compiled, not run on hardware
//
#include <hip/hip_runtime.h>
#include <hip/hip_bf16.h>

typedef __attribute__((ext_vector_type(16))) _Float16 v16h;
typedef __attribute__((ext_vector_type(8)))  float    v8f;

#define N_NODES 50000
#define N_EDGES 800000

// ---------------- WMMA fragment helpers (CDNA5 16x16x32 f16, wave32) ---------
// A-matrix 16x32 f16 (MxK), contiguous layout: lane m (0-15)/m+16 holds one
// contiguous 16-half run: k = 16g..16g+15 (slots 0-15), g = lane>>4. With a
// row stride that is a multiple of 8 halves this is
// exactly two 16-byte LDS loads.
__device__ inline v16h frag_A_vec(const _Float16* p, int ld, int lane) {
  const int m = lane & 15, g = (lane >> 4) & 1;
  const _Float16* q = p + m * ld + g * 16;
  union { v16h v; uint4 u[2]; } r;
  r.u[0] = *(const uint4*)(q);        // slots 0-7
  r.u[1] = *(const uint4*)(q + 8);    // slots 8-15
  return r.v;
}

// B fragments are pre-packed in LDS in fragment-major order:
// wf[(frag*32 + lane)*16 + slot], i.e. 32 contiguous bytes per lane.
__device__ inline v16h frag_B_pk(const _Float16* wf, int frag, int lane) {
  const _Float16* q = wf + ((frag * 32 + lane) << 4);
  union { v16h v; uint4 u[2]; } r;
  r.u[0] = *(const uint4*)(q);
  r.u[1] = *(const uint4*)(q + 8);
  return r.v;
}

__device__ inline v8f wmma16(v16h a, v16h b, v8f c) {
  return __builtin_amdgcn_wmma_f32_16x16x32_f16(false, a, false, b,
                                                (short)0, c, false, false);
}

// pack 4 floats -> 4 f16 -> 8-byte LDS store
__device__ inline void st_half4(_Float16* dst, float4 v) {
  union { _Float16 h[4]; uint2 u; } pk;
  pk.h[0] = (_Float16)v.x; pk.h[1] = (_Float16)v.y;
  pk.h[2] = (_Float16)v.z; pk.h[3] = (_Float16)v.w;
  *(uint2*)dst = pk.u;
}

// ---------------- small utility kernels -------------------------------------

__global__ void k_zero(float* __restrict__ p, int n) {
  int i = blockIdx.x * blockDim.x + threadIdx.x;
  if (i < n) p[i] = 0.f;
}

__global__ void k_deg(const int* __restrict__ ei, float* __restrict__ cnt,
                      float* __restrict__ deg, int E) {
  int e = blockIdx.x * blockDim.x + threadIdx.x;
  if (e < E) {
    atomicAdd(&cnt[ei[e]], 1.f);      // row count (segment-mean denom)
    atomicAdd(&deg[ei[E + e]], 1.f);  // col in-degree (GCN D)
  }
}

__global__ void k_dinv(const float* __restrict__ deg, float* __restrict__ dinv,
                       int n) {
  int i = blockIdx.x * blockDim.x + threadIdx.x;
  if (i < n) dinv[i] = rsqrtf(deg[i] + 1.f);  // +1 self-loop, always > 0
}

// x32[v][0..29] = nodes + agg/max(cnt,1) ; x32[v][30..31] = 0 (K padding)
__global__ void k_x(const float* __restrict__ nodes, const float* __restrict__ agg,
                    const float* __restrict__ cnt, float* __restrict__ x32, int n) {
  int i = blockIdx.x * blockDim.x + threadIdx.x;
  if (i < n) {
    int v = i >> 5, d = i & 31;
    float val = 0.f;
    if (d < 30) {
      float c = cnt[v];
      c = c < 1.f ? 1.f : c;
      val = nodes[v * 30 + d] + agg[v * 30 + d] / c;
    }
    x32[i] = val;
  }
}

// acc[v][f] = h[v][f]*dinv[v]^2 + b[f]   (self-loop term + bias)
__global__ void k_init_acc(const float* __restrict__ h, const float* __restrict__ dinv,
                           const float* __restrict__ b, float* __restrict__ acc,
                           int N, int F) {
  int i = blockIdx.x * blockDim.x + threadIdx.x;
  if (i < N * F) {
    int v = i / F, f = i - v * F;
    float d = dinv[v];
    acc[i] = h[i] * d * d + b[f];
  }
}

// acc[col[e]] += h[row[e]] * dinv[row]*dinv[col], float4 per thread
__global__ void k_scatter(const float* __restrict__ h, const float* __restrict__ dinv,
                          const int* __restrict__ ei, float* __restrict__ acc,
                          int E, int F) {
  int idx = blockIdx.x * blockDim.x + threadIdx.x;
  const int q = F >> 2;
  if (idx < E * q) {
    int e = idx / q, f4 = idx - e * q;
    int r = ei[e], c = ei[E + e];
    float wgt = dinv[r] * dinv[c];
    const float4 hv = *(const float4*)(h + (size_t)r * F + f4 * 4);
    float* d = acc + (size_t)c * F + f4 * 4;
    atomicAdd(d + 0, hv.x * wgt);
    atomicAdd(d + 1, hv.y * wgt);
    atomicAdd(d + 2, hv.z * wgt);
    atomicAdd(d + 3, hv.w * wgt);
  }
}

__global__ void k_relu(float* __restrict__ p, int n) {
  int i = blockIdx.x * blockDim.x + threadIdx.x;
  if (i < n) p[i] = fmaxf(p[i], 0.f);
}

__global__ void k_colsum(const float* __restrict__ x, float* __restrict__ out,
                         int N, int F) {
  __shared__ float sm[256];
  const int f = blockIdx.x;
  float s = 0.f;
  for (int v = threadIdx.x; v < N; v += 256) s += x[(size_t)v * F + f];
  sm[threadIdx.x] = s;
  __syncthreads();
  for (int o = 128; o > 0; o >>= 1) {
    if ((int)threadIdx.x < o) sm[threadIdx.x] += sm[threadIdx.x + o];
    __syncthreads();
  }
  if (threadIdx.x == 0) out[f] = sm[0];
}

__global__ void k_head(const float* __restrict__ gsum, const float* __restrict__ act,
                       const float* __restrict__ sW, const float* __restrict__ sb,
                       const float* __restrict__ aW, const float* __restrict__ ab,
                       const float* __restrict__ q1W, const float* __restrict__ q1b,
                       const float* __restrict__ qoW, const float* __restrict__ qob,
                       float* __restrict__ out, int N) {
  __shared__ float g[64], cc[128], red[64];
  const int j = threadIdx.x;
  g[j] = gsum[j] / (float)N;
  __syncthreads();
  float accs = sb[j];
  for (int i = 0; i < 64; ++i) accs += g[i] * sW[i * 64 + j];
  float acca = ab[j];
  for (int i = 0; i < 30; ++i) acca += act[i] * aW[i * 64 + j];
  cc[j] = fmaxf(accs, 0.f);
  cc[64 + j] = fmaxf(acca, 0.f);
  __syncthreads();
  float accq = q1b[j];
  for (int k = 0; k < 128; ++k) accq += cc[k] * q1W[k * 64 + j];
  red[j] = fmaxf(accq, 0.f) * qoW[j];
  __syncthreads();
  for (int o = 32; o > 0; o >>= 1) {
    if (j < o) red[j] += red[j + o];
    __syncthreads();
  }
  if (j == 0) out[0] = red[0] + qob[0];
}

// ---------------- WMMA kernel 1: fused edge MLP + scatter-sum ---------------
// 8 waves/block, 16 edges per wave; weights packed once per block into
// fragment-major LDS so each B fragment is 2x ds_load_b128.
__global__ __launch_bounds__(256) void k_edge_mlp(
    const float* __restrict__ eattr, const int* __restrict__ ei,
    const float* __restrict__ e1W, const float* __restrict__ e1b,
    const float* __restrict__ e2W, const float* __restrict__ e2b,
    float* __restrict__ agg, int E) {
  __shared__ __align__(16) _Float16 wf1[8 * 512];    // e1_W, 8 n-frags (K 16->32 pad)
  __shared__ __align__(16) _Float16 wf2[8 * 512];    // e2_W, 4 kc x 2 nt frags (N 30->32)
  __shared__ float b1s[128];
  __shared__ float b2s[32];
  __shared__ __align__(16) _Float16 at[8][16 * 40];  // per-wave edge tile, K-padded
  __shared__ __align__(16) _Float16 hid[8][16 * 128];

  const int tid = threadIdx.x, lane = tid & 31, w = tid >> 5;

  // pack e1_W [16x128] -> frag f = nt: (lane l, slot s) <- k=s+16*(l>>4), n=f*16+(l&15)
  for (int i = tid; i < 8 * 512; i += 256) {
    int f = i >> 9, ls = i & 511, l = ls >> 4, s = ls & 15;
    int k = s + 16 * (l >> 4);
    int n = f * 16 + (l & 15);
    wf1[i] = (k < 16) ? (_Float16)e1W[k * 128 + n] : (_Float16)0.f;
  }
  // pack e2_W [128x30->32] -> frag f = kc*2+nt
  for (int i = tid; i < 8 * 512; i += 256) {
    int f = i >> 9, ls = i & 511, l = ls >> 4, s = ls & 15;
    int kc = f >> 1, nt = f & 1;
    int k = kc * 32 + s + 16 * (l >> 4);
    int n = nt * 16 + (l & 15);
    wf2[i] = (n < 30) ? (_Float16)e2W[k * 30 + n] : (_Float16)0.f;
  }
  if (tid < 128) b1s[tid] = e1b[tid];
  if (tid < 32)  b2s[tid] = (tid < 30) ? e2b[tid] : 0.f;

  // stage 16x16 f32 edge tile -> f16 LDS (stride 40), float4 -> half4
  const int e0 = (blockIdx.x * 8 + w) * 16;
  for (int j = lane; j < 64; j += 32) {
    int r = j >> 2, c4 = j & 3;
    float4 v = *(const float4*)(eattr + (size_t)(e0 + r) * 16 + c4 * 4);
    st_half4(&at[w][r * 40 + c4 * 4], v);
  }
  {  // zero K pad 16..31 (one uint4 per lane)
    int r = lane >> 1, q = lane & 1;
    uint4 z = {0u, 0u, 0u, 0u};
    *(uint4*)(&at[w][r * 40 + 16 + q * 8]) = z;
  }
  __syncthreads();

  const int nlo = lane & 15, mb = (lane >> 4) * 8;

  // stage 1: [16x32] @ [32x128]
  v16h a1 = frag_A_vec(at[w], 40, lane);
#pragma unroll
  for (int n0 = 0; n0 < 8; ++n0) {
    v16h b = frag_B_pk(wf1, n0, lane);
    v8f c = {0.f, 0.f, 0.f, 0.f, 0.f, 0.f, 0.f, 0.f};
    c = wmma16(a1, b, c);
    union { v8f v; float f[8]; } u; u.v = c;
    const int n = nlo + n0 * 16;
    const float bias = b1s[n];
#pragma unroll
    for (int r = 0; r < 8; ++r)
      hid[w][(mb + r) * 128 + n] = (_Float16)fmaxf(u.f[r] + bias, 0.f);
  }
  __syncthreads();

  // stage 2: [16x128] @ [128x32], K in 4 chunks
  v16h a2[4];
#pragma unroll
  for (int kc = 0; kc < 4; ++kc) a2[kc] = frag_A_vec(hid[w] + kc * 32, 128, lane);
#pragma unroll
  for (int nt = 0; nt < 2; ++nt) {
    v8f c = {0.f, 0.f, 0.f, 0.f, 0.f, 0.f, 0.f, 0.f};
#pragma unroll
    for (int kc = 0; kc < 4; ++kc)
      c = wmma16(a2[kc], frag_B_pk(wf2, kc * 2 + nt, lane), c);
    const int n = nlo + nt * 16;
    if (n < 30) {
      union { v8f v; float f[8]; } u; u.v = c;
      const float bias = b2s[n];
#pragma unroll
      for (int r = 0; r < 8; ++r) {
        const int e = e0 + mb + r;
        if (e < E) {
          float val = fmaxf(u.f[r] + bias, 0.f);
          atomicAdd(&agg[(size_t)ei[e] * 30 + n], val);
        }
      }
    }
  }
}

// ---------------- WMMA kernel 2: h1 = f16(x32[N,32]) @ f16(g1_W[30,128]) ----
__global__ __launch_bounds__(128) void k_gemm1(
    const float* __restrict__ x32, const float* __restrict__ g1W,
    float* __restrict__ h1, int Ntiles) {
  __shared__ __align__(16) _Float16 wf[8 * 512];
  __shared__ __align__(16) _Float16 xt[4][16 * 40];
  const int tid = threadIdx.x, lane = tid & 31, w = tid >> 5;
  for (int i = tid; i < 8 * 512; i += 128) {
    int f = i >> 9, ls = i & 511, l = ls >> 4, s = ls & 15;
    int k = s + 16 * (l >> 4);
    int n = f * 16 + (l & 15);
    wf[i] = (k < 30) ? (_Float16)g1W[k * 128 + n] : (_Float16)0.f;
  }
  const int t = blockIdx.x * 4 + w;
  const bool ok = t < Ntiles;
  const int v0 = t * 16;
  if (ok) {
    for (int j = lane; j < 128; j += 32) {  // 16 rows x 8 float4 (K padded in x32)
      int r = j >> 3, c4 = j & 7;
      float4 v = *(const float4*)(x32 + (size_t)(v0 + r) * 32 + c4 * 4);
      st_half4(&xt[w][r * 40 + c4 * 4], v);
    }
  }
  __syncthreads();
  if (!ok) return;
  v16h a = frag_A_vec(xt[w], 40, lane);
  const int nlo = lane & 15, mb = (lane >> 4) * 8;
#pragma unroll
  for (int n0 = 0; n0 < 8; ++n0) {
    v8f c = {0.f, 0.f, 0.f, 0.f, 0.f, 0.f, 0.f, 0.f};
    c = wmma16(a, frag_B_pk(wf, n0, lane), c);
    union { v8f v; float f[8]; } u; u.v = c;
    const int n = nlo + n0 * 16;
#pragma unroll
    for (int r = 0; r < 8; ++r)
      h1[(size_t)(v0 + mb + r) * 128 + n] = u.f[r];
  }
}

// ---------------- WMMA kernel 3: h2 = f16(x1[N,128]) @ f16(g2_W[128,64]) ----
__global__ __launch_bounds__(128) void k_gemm2(
    const float* __restrict__ x1, const float* __restrict__ g2W,
    float* __restrict__ h2, int Ntiles) {
  __shared__ __align__(16) _Float16 wf[16 * 512];     // 4 kc x 4 nt frags
  __shared__ __align__(16) _Float16 xt[4][16 * 136];
  const int tid = threadIdx.x, lane = tid & 31, w = tid >> 5;
  for (int i = tid; i < 16 * 512; i += 128) {
    int f = i >> 9, ls = i & 511, l = ls >> 4, s = ls & 15;
    int kc = f >> 2, nt = f & 3;
    int k = kc * 32 + s + 16 * (l >> 4);
    int n = nt * 16 + (l & 15);
    wf[i] = (_Float16)g2W[k * 64 + n];
  }
  const int t = blockIdx.x * 4 + w;
  const bool ok = t < Ntiles;
  const int v0 = t * 16;
  if (ok) {
    for (int j = lane; j < 512; j += 32) {  // 16 rows x 32 float4
      int r = j >> 5, c4 = j & 31;
      float4 v = *(const float4*)(x1 + (size_t)(v0 + r) * 128 + c4 * 4);
      st_half4(&xt[w][r * 136 + c4 * 4], v);
    }
  }
  __syncthreads();
  if (!ok) return;
  v16h a[4];
#pragma unroll
  for (int kc = 0; kc < 4; ++kc) a[kc] = frag_A_vec(xt[w] + kc * 32, 136, lane);
  const int nlo = lane & 15, mb = (lane >> 4) * 8;
#pragma unroll
  for (int nt = 0; nt < 4; ++nt) {
    v8f c = {0.f, 0.f, 0.f, 0.f, 0.f, 0.f, 0.f, 0.f};
#pragma unroll
    for (int kc = 0; kc < 4; ++kc)
      c = wmma16(a[kc], frag_B_pk(wf, kc * 4 + nt, lane), c);
    union { v8f v; float f[8]; } u; u.v = c;
    const int n = nlo + nt * 16;
#pragma unroll
    for (int r = 0; r < 8; ++r)
      h2[(size_t)(v0 + mb + r) * 64 + n] = u.f[r];
  }
}

// ---------------- launch ----------------------------------------------------

static inline int cdiv_i(long a, long b) { return (int)((a + b - 1) / b); }

extern "C" void kernel_launch(void* const* d_in, const int* in_sizes, int n_in,
                              void* d_out, int out_size, void* d_ws, size_t ws_size,
                              hipStream_t stream) {
  (void)in_sizes; (void)n_in; (void)out_size; (void)ws_size;
  const float* nodes = (const float*)d_in[0];
  const float* eattr = (const float*)d_in[1];
  const float* act   = (const float*)d_in[2];
  const int*   ei    = (const int*)d_in[3];
  const float* e1W = (const float*)d_in[4];  const float* e1b = (const float*)d_in[5];
  const float* e2W = (const float*)d_in[6];  const float* e2b = (const float*)d_in[7];
  const float* g1W = (const float*)d_in[8];  const float* g1b = (const float*)d_in[9];
  const float* g2W = (const float*)d_in[10]; const float* g2b = (const float*)d_in[11];
  const float* sW  = (const float*)d_in[12]; const float* sb  = (const float*)d_in[13];
  const float* aW  = (const float*)d_in[14]; const float* ab  = (const float*)d_in[15];
  const float* q1W = (const float*)d_in[16]; const float* q1b = (const float*)d_in[17];
  const float* qoW = (const float*)d_in[18]; const float* qob = (const float*)d_in[19];
  float* out = (float*)d_out;
  float* ws  = (float*)d_ws;

  const int N = N_NODES, E = N_EDGES;

  // workspace layout (floats); h2 aliases h1 (h1 dead after conv1 scatter)
  float* cnt  = ws + 0;                       // 50000
  float* deg  = ws + 50000;                   // 50000
  float* dinv = ws + 100000;                  // 50000
  float* gsum = ws + 150000;                  // 64
  float* agg  = ws + 150080;                  // N*30
  float* x32  = agg  + (size_t)N * 30;        // N*32 (K-padded)
  float* h1   = x32  + (size_t)N * 32;        // N*128
  float* acc1 = h1   + (size_t)N * 128;       // N*128  (becomes x1)
  float* h2   = h1;                           // N*64 within h1 region
  float* acc2 = acc1 + (size_t)N * 128;       // N*64   (becomes x2)

  k_zero<<<cdiv_i(100000, 256), 256, 0, stream>>>(ws, 100000);          // cnt+deg
  k_zero<<<cdiv_i((long)N * 30, 256), 256, 0, stream>>>(agg, N * 30);
  k_deg<<<cdiv_i(E, 256), 256, 0, stream>>>(ei, cnt, deg, E);

  // fused edge MLP + scatter-sum (WMMA), 8 tiles/block
  k_edge_mlp<<<E / 128, 256, 0, stream>>>(eattr, ei, e1W, e1b, e2W, e2b, agg, E);

  k_dinv<<<cdiv_i(N, 256), 256, 0, stream>>>(deg, dinv, N);
  k_x<<<cdiv_i((long)N * 32, 256), 256, 0, stream>>>(nodes, agg, cnt, x32, N * 32);

  // GCN conv 1
  const int tiles = N / 16;                                             // 3125
  k_gemm1<<<cdiv_i(tiles, 4), 128, 0, stream>>>(x32, g1W, h1, tiles);
  k_init_acc<<<cdiv_i((long)N * 128, 256), 256, 0, stream>>>(h1, dinv, g1b, acc1, N, 128);
  k_scatter<<<cdiv_i((long)E * 32, 256), 256, 0, stream>>>(h1, dinv, ei, acc1, E, 128);
  k_relu<<<cdiv_i((long)N * 128, 256), 256, 0, stream>>>(acc1, N * 128);

  // GCN conv 2
  k_gemm2<<<cdiv_i(tiles, 4), 128, 0, stream>>>(acc1, g2W, h2, tiles);
  k_init_acc<<<cdiv_i((long)N * 64, 256), 256, 0, stream>>>(h2, dinv, g2b, acc2, N, 64);
  k_scatter<<<cdiv_i((long)E * 16, 256), 256, 0, stream>>>(h2, dinv, ei, acc2, E, 64);
  k_relu<<<cdiv_i((long)N * 64, 256), 256, 0, stream>>>(acc2, N * 64);

  // pool + head
  k_colsum<<<64, 256, 0, stream>>>(acc2, gsum, N, 64);
  k_head<<<1, 64, 0, stream>>>(gsum, act, sW, sb, aW, ab, q1W, q1b, qoW, qob, out, N);
}
